// CAGAM_75831942578813
// MI455X (gfx1250) — matrix operand order repaired and checked by reading, not compiled, not run on hardware
//
#include <hip/hip_runtime.h>
#include <cstdint>
#include <cstddef>

typedef _Float16 half_t;
typedef __attribute__((ext_vector_type(16))) _Float16 v16h;
typedef __attribute__((ext_vector_type(8)))  _Float16 v8h;
typedef __attribute__((ext_vector_type(8)))  float    v8f;

#define NB   8
#define NC   256
#define NHW  1024
#define NL   1024
#define NDI  512
#define NM   8192      /* NB * NL */
#define NDS  16
#define EPSV 1e-5f

/* Async-to-LDS path availability (gfx1250 builtins; guarded so compile is safe) */
#if defined(__has_builtin)
# if __has_builtin(__builtin_amdgcn_global_load_async_to_lds_b32) && \
     __has_builtin(__builtin_amdgcn_s_wait_asynccnt)
#  define CAGAM_ASYNC 1
# endif
#endif
#ifndef CAGAM_ASYNC
# define CAGAM_ASYNC 0
#endif

#if CAGAM_ASYNC
typedef __attribute__((address_space(1))) int* cagam_gptr;
typedef __attribute__((address_space(3))) int* cagam_lptr;
#endif

/* ---------------------------------------------------------------------- */
/* WMMA GEMM:  C[m,n] = sum_k A[m,k] * W[n,k]   (x @ W.T, W row-major N,K)*/
/* One wave computes a 32x32 tile = 2x2 v_wmma_f32_16x16x32_f16 tiles.    */
/* EDGE=true only for the N=48 x-proj GEMM (predicated B load + store).   */
/* ---------------------------------------------------------------------- */

__device__ __forceinline__ v16h cagam_ld_a(const half_t* A, int lda, int row,
                                           int k0, int lane) {
  int r  = row + (lane & 15);
  int kh = k0 + ((lane >> 4) << 3);          /* +0 or +8  (A 16x32 layout) */
  const half_t* p = A + (size_t)r * lda + kh;
  v8h x0 = *(const v8h*)p;
  v8h x1 = *(const v8h*)(p + 16);
  v16h o;
#pragma unroll
  for (int i = 0; i < 8; ++i) { o[i] = x0[i]; o[i + 8] = x1[i]; }
  return o;
}

template <bool EDGE>
__device__ __forceinline__ v16h cagam_ld_b(const half_t* W, int ldw, int col,
                                           int k0, int lane, int N) {
  int n  = col + (lane & 15);
  int kh = k0 + ((lane >> 4) << 3);          /* +0 or +8  (B 32x16 layout) */
  v16h o;
  if (EDGE && n >= N) {
#pragma unroll
    for (int i = 0; i < 16; ++i) o[i] = (half_t)0.f;
    return o;
  }
  const half_t* p = W + (size_t)n * ldw + kh;
  v8h x0 = *(const v8h*)p;
  v8h x1 = *(const v8h*)(p + 16);
#pragma unroll
  for (int i = 0; i < 8; ++i) { o[i] = x0[i]; o[i + 8] = x1[i]; }
  return o;
}

template <bool EDGE>
__global__ void cagam_gemm_wmma(const half_t* __restrict__ A,
                                const half_t* __restrict__ Wm,
                                float* __restrict__ Cm,
                                int M, int N, int K, int lda, int ldb, int ldc,
                                long long sA, long long sB, long long sC) {
  int lane = threadIdx.x & 31;
  int wv   = threadIdx.x >> 5;
  int tilesN = (N + 31) >> 5;
  int tilesM = (M + 31) >> 5;
  int gt = blockIdx.x * (blockDim.x >> 5) + wv;
  if (gt >= tilesM * tilesN) return;        /* wave-uniform exit */
  int tm = (gt / tilesN) * 32;
  int tn = (gt % tilesN) * 32;
  const half_t* Ab = A  + (size_t)sA * blockIdx.y;
  const half_t* Bb = Wm + (size_t)sB * blockIdx.y;
  float*        Cb = Cm + (size_t)sC * blockIdx.y;

  v8f a00 = {}, a01 = {}, a10 = {}, a11 = {};
  for (int k0 = 0; k0 < K; k0 += 32) {
    if (k0 + 32 < K) {                       /* stream next K-slab toward L0 */
      __builtin_prefetch((const void*)(Ab + (size_t)(tm + (lane & 15)) * lda + k0 + 32), 0, 1);
      __builtin_prefetch((const void*)(Bb + (size_t)(tn + (lane & 15)) * ldb + k0 + 32), 0, 1);
    }
    v16h fa0 = cagam_ld_a(Ab, lda, tm,      k0, lane);
    v16h fa1 = cagam_ld_a(Ab, lda, tm + 16, k0, lane);
    v16h fb0 = cagam_ld_b<EDGE>(Bb, ldb, tn,      k0, lane, N);
    v16h fb1 = cagam_ld_b<EDGE>(Bb, ldb, tn + 16, k0, lane, N);
    a00 = __builtin_amdgcn_wmma_f32_16x16x32_f16(false, fa0, false, fb0, (short)0, a00, false, false);
    a01 = __builtin_amdgcn_wmma_f32_16x16x32_f16(false, fa0, false, fb1, (short)0, a01, false, false);
    a10 = __builtin_amdgcn_wmma_f32_16x16x32_f16(false, fa1, false, fb0, (short)0, a10, false, false);
    a11 = __builtin_amdgcn_wmma_f32_16x16x32_f16(false, fa1, false, fb1, (short)0, a11, false, false);
  }
  int cl = lane & 15, rh = (lane >> 4) * 8;
#pragma unroll
  for (int i = 0; i < 8; ++i) {
    int m0 = tm + rh + i, m1 = tm + 16 + rh + i;
    int n0 = tn + cl,     n1 = tn + 16 + cl;
    if (!EDGE || n0 < N) { Cb[(size_t)m0 * ldc + n0] = a00[i]; Cb[(size_t)m1 * ldc + n0] = a10[i]; }
    if (!EDGE || n1 < N) { Cb[(size_t)m0 * ldc + n1] = a01[i]; Cb[(size_t)m1 * ldc + n1] = a11[i]; }
  }
}

/* ---------------------------------------------------------------------- */
/* Elementwise / reduction kernels                                        */
/* ---------------------------------------------------------------------- */

__global__ void cagam_f2h(const float* __restrict__ in, half_t* __restrict__ out, int n) {
  int i = blockIdx.x * 256 + threadIdx.x;
  if (i < n) out[i] = (half_t)in[i];
}

/* LayerNorm over C; x:(B,C,H,W) -> xn_h:(B*L, C) fp16 */
__global__ void cagam_ln(const float* __restrict__ x, const float* __restrict__ g,
                         const float* __restrict__ bt, half_t* __restrict__ out) {
  int bl = blockIdx.x;                 /* 0..NM-1 */
  int b  = bl >> 10, l = bl & 1023;
  int c  = threadIdx.x;                /* 256 threads = channels */
  float v = x[((size_t)b * NC + c) * NHW + l];
  __shared__ float red[256];
  red[c] = v; __syncthreads();
  for (int s = 128; s > 0; s >>= 1) { if (c < s) red[c] += red[c + s]; __syncthreads(); }
  float mean = red[0] * (1.f / NC);
  __syncthreads();
  float d = v - mean;
  red[c] = d * d; __syncthreads();
  for (int s = 128; s > 0; s >>= 1) { if (c < s) red[c] += red[c + s]; __syncthreads(); }
  float rs = rsqrtf(red[0] * (1.f / NC) + EPSV);
  out[(size_t)bl * NC + c] = (half_t)(d * rs * g[c] + bt[c]);
}

/* gather x (cols 0..511 of xz) into direction layout (seq,t,d) */
__global__ void cagam_gather(const float* __restrict__ xz, float* __restrict__ xd, int dir) {
  size_t i = (size_t)blockIdx.x * 256 + threadIdx.x;   /* over NM*NDI */
  if (i >= (size_t)NM * NDI) return;
  int d = (int)(i & (NDI - 1));
  size_t bl = i >> 9;
  int b = (int)(bl >> 10), l = (int)(bl & 1023);
  size_t row;
  if      (dir == 0) row = bl;
  else if (dir == 1) row = (size_t)b * NL + (NL - 1 - l);
  else               row = (size_t)(b * 4 + (l & 3)) * 256 + (l >> 2);
  xd[row * NDI + d] = xz[bl * 1024 + d];
}

/* causal depthwise conv (DCONV=4) + bias + SiLU; writes f32 + f16 */
__global__ void cagam_conv(const float* __restrict__ xd, const float* __restrict__ cw,
                           const float* __restrict__ cb, float* __restrict__ xc,
                           half_t* __restrict__ xch, int Ls) {
  size_t i = (size_t)blockIdx.x * 256 + threadIdx.x;   /* over NM*NDI */
  if (i >= (size_t)NM * NDI) return;
  int d = (int)(i & (NDI - 1));
  size_t st = i >> 9;
  int t = (int)(st % Ls);
  size_t s0 = st - t;
  float acc = cb[d];
#pragma unroll
  for (int j = 0; j < 4; ++j) {
    int ti = t - 3 + j;
    if (ti >= 0) acc += xd[(s0 + ti) * NDI + d] * cw[d * 4 + j];
  }
  float v = acc / (1.f + __expf(-acc));    /* SiLU */
  xc[i] = v;
  xch[i] = (half_t)v;
}

/* dt = softplus(x_dbl[:, :16] @ dt_W.T + dt_b)   (K=16, scalar path) */
__global__ void cagam_dt(const float* __restrict__ xdbl, const float* __restrict__ dtW,
                         const float* __restrict__ dtb, float* __restrict__ dt) {
  int row = blockIdx.x;                  /* 0..NM-1 */
  __shared__ float xr[16];
  if (threadIdx.x < 16) xr[threadIdx.x] = xdbl[(size_t)row * 48 + threadIdx.x];
  __syncthreads();
  for (int d = threadIdx.x; d < NDI; d += 256) {
    float s = dtb[d];
#pragma unroll
    for (int r = 0; r < 16; ++r) s += xr[r] * dtW[d * 16 + r];
    dt[(size_t)row * NDI + d] = (s > 20.f) ? s : log1pf(__expf(s));
  }
}

/* selective scan: one thread per channel, state h[16] in registers.      */
/* Async path: wave0 streams step t+1's B/C row into double-buffered LDS  */
/* via global_load_async_to_lds while all waves compute step t.           */
__global__ void cagam_scan(const float* __restrict__ xc, const float* __restrict__ dt,
                           const float* __restrict__ xdbl, const float* __restrict__ Alog,
                           const float* __restrict__ Dp, float* __restrict__ yb, int Ls) {
  int seq = blockIdx.x;
  int d   = blockIdx.y * 256 + threadIdx.x;
  float Ar[NDS];
#pragma unroll
  for (int n = 0; n < NDS; ++n) Ar[n] = -__expf(Alog[d * NDS + n]);
  float Dd = Dp[d];
  float h[NDS];
#pragma unroll
  for (int n = 0; n < NDS; ++n) h[n] = 0.f;
  size_t base = (size_t)seq * Ls;
  const float* grow = xdbl + base * 48 + 16;   /* row t at grow + t*48 */

#if CAGAM_ASYNC
  __shared__ float bc2[2][32];
  if (threadIdx.x < 32)
    __builtin_amdgcn_global_load_async_to_lds_b32(
        (cagam_gptr)(grow + threadIdx.x),
        (cagam_lptr)&bc2[0][threadIdx.x], 0, 0);
  for (int t = 0; t < Ls; ++t) {
    if (threadIdx.x < 32) __builtin_amdgcn_s_wait_asynccnt(0);
    __syncthreads();
    if (threadIdx.x < 32 && t + 1 < Ls)
      __builtin_amdgcn_global_load_async_to_lds_b32(
          (cagam_gptr)(grow + (size_t)(t + 1) * 48 + threadIdx.x),
          (cagam_lptr)&bc2[(t + 1) & 1][threadIdx.x], 0, 0);
    const float* bc = bc2[t & 1];
    float dtv = dt[(base + t) * NDI + d];
    float u   = xc[(base + t) * NDI + d];
    float du  = dtv * u;
    float y   = 0.f;
#pragma unroll
    for (int n = 0; n < NDS; ++n) {
      h[n] = __expf(dtv * Ar[n]) * h[n] + du * bc[n];
      y   += h[n] * bc[16 + n];
    }
    yb[(base + t) * NDI + d] = y + u * Dd;
  }
#else
  for (int t = 0; t < Ls; ++t) {
    float bcv[32];
    const float4* r = (const float4*)(grow + (size_t)t * 48);
#pragma unroll
    for (int q = 0; q < 8; ++q) {
      float4 v4 = r[q];
      bcv[4 * q] = v4.x; bcv[4 * q + 1] = v4.y; bcv[4 * q + 2] = v4.z; bcv[4 * q + 3] = v4.w;
    }
    float dtv = dt[(base + t) * NDI + d];
    float u   = xc[(base + t) * NDI + d];
    float du  = dtv * u;
    float y   = 0.f;
#pragma unroll
    for (int n = 0; n < NDS; ++n) {
      h[n] = __expf(dtv * Ar[n]) * h[n] + du * bcv[n];
      y   += h[n] * bcv[16 + n];
    }
    yb[(base + t) * NDI + d] = y + u * Dd;
  }
#endif
}

/* ysum[b,l,d] (op)= y_dir[map(b,l),d] * silu(z[b,l,d]) ; dir0 writes     */
__global__ void cagam_scatter(const float* __restrict__ yb, const float* __restrict__ xz,
                              float* __restrict__ ysum, int dir) {
  size_t i = (size_t)blockIdx.x * 256 + threadIdx.x;   /* over NM*NDI */
  if (i >= (size_t)NM * NDI) return;
  int d = (int)(i & (NDI - 1));
  size_t bl = i >> 9;
  int b = (int)(bl >> 10), l = (int)(bl & 1023);
  size_t row;
  if      (dir == 0) row = bl;
  else if (dir == 1) row = (size_t)b * NL + (NL - 1 - l);
  else               row = (size_t)(b * 4 + (l & 3)) * 256 + (l >> 2);
  float z = xz[bl * 1024 + NDI + d];
  float v = yb[row * NDI + d] * (z / (1.f + __expf(-z)));
  if (dir == 0) ysum[i] = v; else ysum[i] += v;
}

/* (B,L,C) f32  ->  (B,C,L) fp16 */
__global__ void cagam_trans_h(const float* __restrict__ in, half_t* __restrict__ out) {
  size_t i = (size_t)blockIdx.x * 256 + threadIdx.x;   /* over B*C*L, out index */
  if (i >= (size_t)NB * NC * NL) return;
  int l = (int)(i & 1023);
  size_t bc = i >> 10;
  int c = (int)(bc & 255), b = (int)(bc >> 8);
  out[i] = (half_t)in[((size_t)b * NL + l) * NC + c];
}

/* BN stats over (B,H,W) per channel; input resT:(B,L,C) */
__global__ void cagam_bn_stats(const float* __restrict__ resT, float* __restrict__ stats) {
  int c = blockIdx.x;
  float s = 0.f, s2 = 0.f;
  for (int i = threadIdx.x; i < NB * NL; i += 256) {
    float v = resT[(size_t)i * NC + c];
    s += v; s2 += v * v;
  }
  __shared__ float r1[256], r2[256];
  r1[threadIdx.x] = s; r2[threadIdx.x] = s2; __syncthreads();
  for (int st = 128; st > 0; st >>= 1) {
    if (threadIdx.x < st) { r1[threadIdx.x] += r1[threadIdx.x + st]; r2[threadIdx.x] += r2[threadIdx.x + st]; }
    __syncthreads();
  }
  if (threadIdx.x == 0) {
    float m = r1[0] * (1.f / (NB * NL));
    float v = r2[0] * (1.f / (NB * NL)) - m * m;
    stats[2 * c] = m;
    stats[2 * c + 1] = rsqrtf(v + EPSV);
  }
}

/* normalize + transpose: resT:(B,L,C) -> out:(B,C,H,W) */
__global__ void cagam_bn_apply(const float* __restrict__ resT, const float* __restrict__ stats,
                               const float* __restrict__ g, const float* __restrict__ bb,
                               float* __restrict__ out) {
  size_t i = (size_t)blockIdx.x * 256 + threadIdx.x;   /* over B*C*L, (b,c,l) */
  if (i >= (size_t)NB * NC * NL) return;
  int l = (int)(i & 1023);
  int c = (int)((i >> 10) & 255);
  int b = (int)(i >> 18);
  float v = resT[((size_t)b * NL + l) * NC + c];
  out[i] = (v - stats[2 * c]) * stats[2 * c + 1] * g[c] + bb[c];
}

__global__ void cagam_head_mean(const float* __restrict__ out, float* __restrict__ xbar) {
  int bc = blockIdx.x;                  /* 0..NB*NC-1 */
  float s = 0.f;
  for (int l = threadIdx.x; l < NHW; l += 256) s += out[(size_t)bc * NHW + l];
  __shared__ float r[256];
  r[threadIdx.x] = s; __syncthreads();
  for (int st = 128; st > 0; st >>= 1) { if (threadIdx.x < st) r[threadIdx.x] += r[threadIdx.x + st]; __syncthreads(); }
  if (threadIdx.x == 0) xbar[bc] = r[0] * (1.f / NHW);
}

__global__ void cagam_head_fc1(const float* __restrict__ xbar, const float* __restrict__ W1,
                               const float* __restrict__ b1, float* __restrict__ h1) {
  int i = blockIdx.x * 256 + threadIdx.x;   /* NB * 2C = 4096 */
  if (i >= NB * 2 * NC) return;
  int b = i >> 9, j = i & 511;
  float s = b1[j];
  for (int c = 0; c < NC; ++c) s += xbar[b * NC + c] * W1[j * NC + c];
  h1[i] = (s > 0.f) ? s : expm1f(s);        /* ELU */
}

__global__ void cagam_head_fc2(const float* __restrict__ h1, const float* __restrict__ W2,
                               const float* __restrict__ b2, float* __restrict__ out, int nout) {
  int i = blockIdx.x * 256 + threadIdx.x;
  if (i >= NB * nout) return;
  int b = i / nout, o = i % nout;
  float s = b2[o];
  for (int k = 0; k < 2 * NC; ++k) s += h1[b * 2 * NC + k] * W2[o * 2 * NC + k];
  out[i] = s;
}

/* ---------------------------------------------------------------------- */
/* Host orchestration                                                     */
/* ---------------------------------------------------------------------- */

struct DirP  { const float *Alog, *D, *convW, *convb, *dtW, *dtb, *xprojW; };
struct MambaP{ DirP b; const float *bn_b, *bn_g; DirP f; const float *inW, *ln_b, *ln_g, *outW; DirP s; };
struct ClsP  { const float *W1, *W2, *b1, *b2; };

struct Ws {
  half_t *inW_h, *xproj_h, *outW_h, *xn_h, *xc_h, *ysum_h, *AfT_h, *BfT_h, *BfLC_h, *dot_h;
  float  *xz, *xd, *xc, *xdbl, *dt, *yb, *ysum, *feat, *dot, *resT, *stats, *xbar, *h1;
};

static inline int cagam_cdiv(long long a, long long b) { return (int)((a + b - 1) / b); }

static void cagam_gemm(const half_t* A, const half_t* W, float* Cm,
                       int M, int N, int K, int lda, int ldb, int ldc,
                       int batch, long long sA, long long sB, long long sC,
                       bool edge, hipStream_t st) {
  int tiles = cagam_cdiv(M, 32) * cagam_cdiv(N, 32);
  dim3 g(cagam_cdiv(tiles, 8), batch);
  if (edge)
    cagam_gemm_wmma<true><<<g, 256, 0, st>>>(A, W, Cm, M, N, K, lda, ldb, ldc, sA, sB, sC);
  else
    cagam_gemm_wmma<false><<<g, 256, 0, st>>>(A, W, Cm, M, N, K, lda, ldb, ldc, sA, sB, sC);
}

static void cagam_feat(const float* x, const MambaP& p, half_t* outT_h, half_t* outLC_h,
                       const Ws& w, hipStream_t st) {
  const size_t NELT = (size_t)NM * NDI;
  int gElt = cagam_cdiv((long long)NELT, 256);

  cagam_ln<<<NM, 256, 0, st>>>(x, p.ln_g, p.ln_b, w.xn_h);
  /* in-projection: (8192x256)@(1024x256)^T -> xz (8192x1024) */
  cagam_gemm(w.xn_h, w.inW_h, w.xz, NM, 2 * NDI, NC, NC, NC, 2 * NDI, 1, 0, 0, 0, false, st);

  const DirP* dirs[3] = { &p.f, &p.b, &p.s };
  for (int dir = 0; dir < 3; ++dir) {
    const DirP& dp = *dirs[dir];
    int Ls = (dir == 2) ? 256 : NL;
    cagam_gather<<<gElt, 256, 0, st>>>(w.xz, w.xd, dir);
    cagam_conv<<<gElt, 256, 0, st>>>(w.xd, dp.convW, dp.convb, w.xc, w.xc_h, Ls);
    /* x-proj: (8192x512)@(48x512)^T -> xdbl (edge tiles in N) */
    cagam_gemm(w.xc_h, w.xproj_h + (size_t)dir * 48 * NDI, w.xdbl,
               NM, 48, NDI, NDI, NDI, 48, 1, 0, 0, 0, true, st);
    cagam_dt<<<NM, 256, 0, st>>>(w.xdbl, dp.dtW, dp.dtb, w.dt);
    dim3 gs((dir == 2) ? 32 : NB, NDI / 256);
    cagam_scan<<<gs, 256, 0, st>>>(w.xc, w.dt, w.xdbl, dp.Alog, dp.D, w.yb, Ls);
    cagam_scatter<<<gElt, 256, 0, st>>>(w.yb, w.xz, w.ysum, dir);
  }
  cagam_f2h<<<gElt, 256, 0, st>>>(w.ysum, w.ysum_h, (int)NELT);
  /* out-projection: (8192x512)@(256x512)^T -> feat (B,L,C) */
  cagam_gemm(w.ysum_h, w.outW_h, w.feat, NM, NC, NDI, NDI, NDI, NC, 1, 0, 0, 0, false, st);
  /* (B,C,L) fp16 for the dot GEMM */
  cagam_trans_h<<<cagam_cdiv((long long)NB * NC * NL, 256), 256, 0, st>>>(w.feat, outT_h);
  /* (B,L,C) fp16 for the res GEMM (second operand only) */
  if (outLC_h)
    cagam_f2h<<<cagam_cdiv((long long)NM * NC, 256), 256, 0, st>>>(w.feat, outLC_h, NM * NC);
}

static void cagam_attention(const float* x1, const float* x2, const MambaP& p,
                            const ClsP& cp, int nout, float* out_feat, float* out_head,
                            const Ws& w, hipStream_t st) {
  /* convert weights once per pass */
  cagam_f2h<<<cagam_cdiv(2 * NDI * NC, 256), 256, 0, st>>>(p.inW, w.inW_h, 2 * NDI * NC);
  const DirP* dirs[3] = { &p.f, &p.b, &p.s };
  for (int dir = 0; dir < 3; ++dir)
    cagam_f2h<<<cagam_cdiv(48 * NDI, 256), 256, 0, st>>>(dirs[dir]->xprojW,
        w.xproj_h + (size_t)dir * 48 * NDI, 48 * NDI);
  cagam_f2h<<<cagam_cdiv(NC * NDI, 256), 256, 0, st>>>(p.outW, w.outW_h, NC * NDI);

  cagam_feat(x1, p, w.AfT_h, nullptr,  w, st);
  cagam_feat(x2, p, w.BfT_h, w.BfLC_h, w, st);

  /* dot[b,c,d] = sum_l Af[c,l] * Bf[d,l]   (batched, BT form) */
  cagam_gemm(w.AfT_h, w.BfT_h, w.dot, NC, NC, NL, NL, NL, NC,
             NB, (long long)NC * NL, (long long)NC * NL, (long long)NC * NC, false, st);
  cagam_f2h<<<cagam_cdiv(NB * NC * NC, 256), 256, 0, st>>>(w.dot, w.dot_h, NB * NC * NC);
  /* resT[b,l,c] = sum_d BfLC[l,d] * dot[c,d]  (batched, BT form again) */
  cagam_gemm(w.BfLC_h, w.dot_h, w.resT, NL, NC, NC, NC, NC, NC,
             NB, (long long)NL * NC, (long long)NC * NC, (long long)NL * NC, false, st);

  cagam_bn_stats<<<NC, 256, 0, st>>>(w.resT, w.stats);
  cagam_bn_apply<<<cagam_cdiv((long long)NB * NC * NL, 256), 256, 0, st>>>(
      w.resT, w.stats, p.bn_g, p.bn_b, out_feat);

  cagam_head_mean<<<NB * NC, 256, 0, st>>>(out_feat, w.xbar);
  cagam_head_fc1<<<cagam_cdiv(NB * 2 * NC, 256), 256, 0, st>>>(w.xbar, cp.W1, cp.b1, w.h1);
  cagam_head_fc2<<<1, 256, 0, st>>>(w.h1, cp.W2, cp.b2, out_head, nout);
}

extern "C" void kernel_launch(void* const* d_in, const int* in_sizes, int n_in,
                              void* d_out, int out_size, void* d_ws, size_t ws_size,
                              hipStream_t stream) {
  (void)in_sizes; (void)n_in; (void)out_size; (void)ws_size;
  const float* verb_x   = (const float*)d_in[0];
  const float* target_x = (const float*)d_in[1];
  const float* tool_x   = (const float*)d_in[2];

  /* params flattened in jax tree order (dict keys sorted) */
  int idx = 3;
  auto rdDir = [&](DirP& dp) {
    dp.Alog  = (const float*)d_in[idx++]; dp.D     = (const float*)d_in[idx++];
    dp.convW = (const float*)d_in[idx++]; dp.convb = (const float*)d_in[idx++];
    dp.dtW   = (const float*)d_in[idx++]; dp.dtb   = (const float*)d_in[idx++];
    dp.xprojW= (const float*)d_in[idx++];
  };
  auto rdMamba = [&](MambaP& m) {
    rdDir(m.b);
    m.bn_b = (const float*)d_in[idx++]; m.bn_g = (const float*)d_in[idx++];
    rdDir(m.f);
    m.inW  = (const float*)d_in[idx++];
    m.ln_b = (const float*)d_in[idx++]; m.ln_g = (const float*)d_in[idx++];
    m.outW = (const float*)d_in[idx++];
    rdDir(m.s);
  };
  auto rdCls = [&](ClsP& c) {
    c.W1 = (const float*)d_in[idx++]; c.W2 = (const float*)d_in[idx++];
    c.b1 = (const float*)d_in[idx++]; c.b2 = (const float*)d_in[idx++];
  };
  MambaP m1, m2; ClsP clsT, clsV;
  rdMamba(m1); rdMamba(m2); rdCls(clsT); rdCls(clsV);

  /* workspace bump allocator */
  char* base = (char*)d_ws;
  size_t off = 0;
  auto allocB = [&](size_t bytes) -> void* {
    void* p = base + off;
    off = (off + bytes + 255) & ~(size_t)255;
    return p;
  };
  Ws w;
  w.inW_h  = (half_t*)allocB((size_t)2 * NDI * NC * 2);
  w.xproj_h= (half_t*)allocB((size_t)3 * 48 * NDI * 2);
  w.outW_h = (half_t*)allocB((size_t)NC * NDI * 2);
  w.xn_h   = (half_t*)allocB((size_t)NM * NC * 2);
  w.xz     = (float*) allocB((size_t)NM * 2 * NDI * 4);
  w.xd     = (float*) allocB((size_t)NM * NDI * 4);
  w.xc     = (float*) allocB((size_t)NM * NDI * 4);
  w.xc_h   = (half_t*)allocB((size_t)NM * NDI * 2);
  w.xdbl   = (float*) allocB((size_t)NM * 48 * 4);
  w.dt     = (float*) allocB((size_t)NM * NDI * 4);
  w.yb     = (float*) allocB((size_t)NM * NDI * 4);
  w.ysum   = (float*) allocB((size_t)NM * NDI * 4);
  w.ysum_h = (half_t*)allocB((size_t)NM * NDI * 2);
  w.feat   = (float*) allocB((size_t)NM * NC * 4);
  w.AfT_h  = (half_t*)allocB((size_t)NB * NC * NL * 2);
  w.BfT_h  = (half_t*)allocB((size_t)NB * NC * NL * 2);
  w.BfLC_h = (half_t*)allocB((size_t)NM * NC * 2);
  w.dot    = (float*) allocB((size_t)NB * NC * NC * 4);
  w.dot_h  = (half_t*)allocB((size_t)NB * NC * NC * 2);
  w.resT   = (float*) allocB((size_t)NM * NC * 4);
  w.stats  = (float*) allocB((size_t)2 * NC * 4);
  w.xbar   = (float*) allocB((size_t)NB * NC * 4);
  w.h1     = (float*) allocB((size_t)NB * 2 * NC * 4);

  /* output layout: verb(48) | v_out(B*C*H*W) | target(80) | t_out(B*C*H*W) */
  float* out       = (float*)d_out;
  float* verb_o    = out;
  float* vfeat_o   = out + NB * 6;
  float* target_o  = vfeat_o + (size_t)NB * NC * NHW;
  float* tfeat_o   = target_o + NB * 10;

  cagam_attention(verb_x,   tool_x, m1, clsV, 6,  vfeat_o, verb_o,   w, stream);
  cagam_attention(target_x, tool_x, m2, clsT, 10, tfeat_o, target_o, w, stream);
}
